// DynamicMaskLayer_60017872994799
// MI455X (gfx1250) — matrix operand order repaired and checked by reading, not compiled, not run on hardware
//
#include <hip/hip_runtime.h>
#include <hip/hip_bf16.h>
#include <stdint.h>

#define N_ROWS 32768
#define N_FEAT 1024
#define K_CHUNK 8192
#define KSTEP 64
#define LDSK 72   // 64 K entries + 8 pad (144B per column, multiple of 16B)

typedef __attribute__((ext_vector_type(16))) __bf16 v16bf;
typedef __attribute__((ext_vector_type(8)))  float  v8f;
typedef __attribute__((ext_vector_type(4)))  unsigned int u32x4;
typedef __attribute__((ext_vector_type(2)))  unsigned int u32x2;
typedef __attribute__((ext_vector_type(4)))  float  f32x4;

__device__ __forceinline__ unsigned int f2bf(float f) {
  union { float f; unsigned int u; } v; v.f = f;
  unsigned int u = v.u;
  unsigned int r = u + 0x7FFFu + ((u >> 16) & 1u);   // round-to-nearest-even
  return r >> 16;
}

// ---------------- kernel 0: zero workspace (G + colsum) ----------------
__global__ __launch_bounds__(256) void zero_ws_kernel(float* __restrict__ ws, int n) {
  for (int i = blockIdx.x * blockDim.x + threadIdx.x; i < n; i += gridDim.x * blockDim.x)
    ws[i] = 0.0f;
}

// ---------------- kernel 1: column sums ----------------
__global__ __launch_bounds__(256)
void colsum_kernel(const float* __restrict__ x, float* __restrict__ colsum) {
  const int c  = threadIdx.x * 4;          // 256 threads cover 1024 cols
  const int r0 = blockIdx.x * 128;
  f32x4 acc = {0.f, 0.f, 0.f, 0.f};
  for (int rr = 0; rr < 128; ++rr)
    acc += *(const f32x4*)(x + (size_t)(r0 + rr) * N_FEAT + c);
  atomicAdd(&colsum[c + 0], acc.x);
  atomicAdd(&colsum[c + 1], acc.y);
  atomicAdd(&colsum[c + 2], acc.z);
  atomicAdd(&colsum[c + 3], acc.w);
}

// ---------------- kernel 2: G = x^T x via bf16 WMMA ----------------
// grid (16,16,4): 64x64 tile of G per block over a K chunk of 8192.
// 128 threads = 4 waves, each wave owns a 32x32 subtile (2x2 WMMA frags).
// K-step 64 per barrier: 8 WMMAs per stage.
__global__ __launch_bounds__(128)
void gram_wmma_kernel(const float* __restrict__ x, float* __restrict__ G) {
  __shared__ __align__(16) unsigned short sA[64 * LDSK];
  __shared__ __align__(16) unsigned short sB[64 * LDSK];

  const int tid  = threadIdx.x;
  const int lane = tid & 31;
  const int wave = tid >> 5;
  const int g    = lane >> 4;    // lane half (ISA fragment layout)
  const int lm   = lane & 15;

  const int i0 = blockIdx.y * 64;        // G rows  -> x columns for A
  const int j0 = blockIdx.x * 64;        // G cols  -> x columns for B
  const int kb = blockIdx.z * K_CHUNK;

  const int m0 = (wave >> 1) * 32;
  const int n0 = (wave & 1) * 32;

  v8f c00 = {}, c01 = {}, c10 = {}, c11 = {};

  // A 16x32 bf16: lane half g reads K chunks {8g..8g+7, 16+8g..23+8g}, row = lm
  const int aOff0 = (m0 + lm) * LDSK;
  const int aOff1 = (m0 + 16 + lm) * LDSK;
  // B 32x16 bf16: lane half g reads K chunks {8g..8g+7, 16+8g..23+8g}, col = lm
  const int bOff0 = (n0 + lm) * LDSK;
  const int bOff1 = (n0 + 16 + lm) * LDSK;
  const int aK0 = 8 * g, aK1 = 16 + 8 * g;
  const int bK0 = 8 * g, bK1 = 16 + 8 * g;

  const int cg = (tid & 15) * 4;         // staging column group (0..60)

  for (int ks = 0; ks < K_CHUNK; ks += KSTEP) {
    const int k0 = kb + ks;
    __syncthreads();
    // Stage 64 K-rows x 64 cols for both slabs: 4x4 register tiles,
    // global f32x4 loads -> packed bf16 -> ds_store_b64 (K contiguous).
    #pragma unroll
    for (int it = 0; it < 2; ++it) {
      const int rg = (tid >> 4) * 4 + it * 32;   // K row group
      #pragma unroll
      for (int s = 0; s < 2; ++s) {
        const int cbase = (s ? j0 : i0) + cg;
        unsigned short* sl = s ? sB : sA;
        const size_t rb = (size_t)(k0 + rg) * N_FEAT + cbase;
        f32x4 v0 = *(const f32x4*)(x + rb);
        f32x4 v1 = *(const f32x4*)(x + rb + N_FEAT);
        f32x4 v2 = *(const f32x4*)(x + rb + 2 * N_FEAT);
        f32x4 v3 = *(const f32x4*)(x + rb + 3 * N_FEAT);
        if (ks + KSTEP < K_CHUNK)
          __builtin_prefetch(x + rb + (size_t)KSTEP * N_FEAT, 0, 0);
        #pragma unroll
        for (int j = 0; j < 4; ++j) {
          u32x2 p;
          p.x = f2bf(v0[j]) | (f2bf(v1[j]) << 16);
          p.y = f2bf(v2[j]) | (f2bf(v3[j]) << 16);
          *(u32x2*)(sl + (cg + j) * LDSK + rg) = p;
        }
      }
    }
    __syncthreads();

    #pragma unroll
    for (int kk = 0; kk < 2; ++kk) {
      const int ko = kk * 32;
      union { v16bf v; u32x4 q[2]; } a0, a1, b0, b1;
      a0.q[0] = *(const u32x4*)(sA + aOff0 + ko + aK0);
      a0.q[1] = *(const u32x4*)(sA + aOff0 + ko + aK1);
      a1.q[0] = *(const u32x4*)(sA + aOff1 + ko + aK0);
      a1.q[1] = *(const u32x4*)(sA + aOff1 + ko + aK1);
      b0.q[0] = *(const u32x4*)(sB + bOff0 + ko + bK0);
      b0.q[1] = *(const u32x4*)(sB + bOff0 + ko + bK1);
      b1.q[0] = *(const u32x4*)(sB + bOff1 + ko + bK0);
      b1.q[1] = *(const u32x4*)(sB + bOff1 + ko + bK1);

      c00 = __builtin_amdgcn_wmma_f32_16x16x32_bf16(false, a0.v, false, b0.v, (short)0, c00, false, false);
      c01 = __builtin_amdgcn_wmma_f32_16x16x32_bf16(false, a0.v, false, b1.v, (short)0, c01, false, false);
      c10 = __builtin_amdgcn_wmma_f32_16x16x32_bf16(false, a1.v, false, b0.v, (short)0, c10, false, false);
      c11 = __builtin_amdgcn_wmma_f32_16x16x32_bf16(false, a1.v, false, b1.v, (short)0, c11, false, false);
    }
  }

  // C/D 16x16 f32 layout: VGPR r, lanes 0-15 -> M=r, lanes 16-31 -> M=8+r, N=lm
  const int colBase = j0 + n0 + lm;
  #pragma unroll
  for (int r = 0; r < 8; ++r) {
    const int rowM = 8 * g + r;
    atomicAdd(&G[(size_t)(i0 + m0 + rowM) * N_FEAT + colBase],           c00[r]);
    atomicAdd(&G[(size_t)(i0 + m0 + rowM) * N_FEAT + colBase + 16],      c01[r]);
    atomicAdd(&G[(size_t)(i0 + m0 + 16 + rowM) * N_FEAT + colBase],      c10[r]);
    atomicAdd(&G[(size_t)(i0 + m0 + 16 + rowM) * N_FEAT + colBase + 16], c11[r]);
  }
}

// ---------------- kernel 3: greedy mask via LDS bit-matrix + single-wave scan ----
// Phase 1 (all 32 waves): adj[i][j] = (|corr_ij| > 0.9) as 1024x1024 bitmatrix
// in dynamic LDS (128KB), built with __ballot. Phase 2 (wave 0 only): the
// sequential suppression scan entirely in registers: lane L owns mask bits for
// columns [32L, 32L+32) -> per step one shfl broadcast + one ds word, no barriers.
__global__ __launch_bounds__(1024)
void mask_kernel(const float* __restrict__ G, const float* __restrict__ colsum,
                 float* __restrict__ mask) {
  extern __shared__ unsigned int adj[];          // N_FEAT * 32 words
  __shared__ float smu[N_FEAT];
  __shared__ float sdiag[N_FEAT];
  const int tid  = threadIdx.x;
  const int lane = tid & 31;
  const int w    = tid >> 5;
  const float Nf = (float)N_ROWS;
  {
    const float mu = colsum[tid] * (1.0f / Nf);
    smu[tid]   = mu;
    sdiag[tid] = G[(size_t)tid * N_FEAT + tid] - Nf * mu * mu;   // C_jj
  }
  __syncthreads();
  for (int i = w; i < N_FEAT; i += 32) {
    const float mui = smu[i];
    const float di  = sdiag[i];
    for (int c = 0; c < N_FEAT / 32; ++c) {
      const int j = c * 32 + lane;
      const float Cij = G[(size_t)i * N_FEAT + j] - Nf * mui * smu[j];
      // |corr| > 0.9  <=>  Cij^2 > 0.81 * Cii * Cjj
      const bool pred = (Cij * Cij) > (0.81f * di * sdiag[j]);
      const unsigned word = (unsigned)__ballot(pred);
      if (lane == 0) adj[i * 32 + c] = word;
    }
  }
  __syncthreads();
  if (w == 0) {
    unsigned m = 0xFFFFFFFFu;                    // lane owns cols [32*lane, 32*lane+32)
    for (int i = 0; i < N_FEAT; ++i) {
      const int ow = i >> 5;
      const unsigned alive = ((unsigned)__shfl((int)m, ow, 32) >> (i & 31)) & 1u;
      if (alive) {                               // uniform branch
        m &= ~adj[i * 32 + lane];                // suppress correlated cols
        if (lane == ow) m |= (1u << (i & 31));   // keep feature i itself
      }
    }
    #pragma unroll
    for (int b = 0; b < 32; ++b)
      mask[lane * 32 + b] = ((m >> b) & 1u) ? 1.0f : 0.0f;
  }
}

// ---------------- kernel 4: out = x * mask (broadcast over rows) ----------------
__global__ __launch_bounds__(256)
void scale_kernel(const float* __restrict__ x, const float* __restrict__ mask,
                  float* __restrict__ out) {
  const size_t n4 = (size_t)N_ROWS * N_FEAT / 4;
  const f32x4* x4 = (const f32x4*)x;
  const f32x4* m4 = (const f32x4*)mask;
  f32x4* o4 = (f32x4*)out;
  for (size_t p = (size_t)blockIdx.x * blockDim.x + threadIdx.x; p < n4;
       p += (size_t)gridDim.x * blockDim.x) {
    f32x4 v = x4[p];
    f32x4 m = m4[p & (N_FEAT / 4 - 1)];
    o4[p] = v * m;
  }
}

extern "C" void kernel_launch(void* const* d_in, const int* in_sizes, int n_in,
                              void* d_out, int out_size, void* d_ws, size_t ws_size,
                              hipStream_t stream) {
  (void)in_sizes; (void)n_in; (void)out_size; (void)ws_size;
  const float* x = (const float*)d_in[0];
  float* out = (float*)d_out;

  float* G      = (float*)d_ws;                       // 1024*1024 f32
  float* colsum = G + (size_t)N_FEAT * N_FEAT;        // 1024 f32
  float* mask   = colsum + N_FEAT;                    // 1024 f32

  zero_ws_kernel<<<512, 256, 0, stream>>>(G, N_FEAT * N_FEAT + N_FEAT);
  colsum_kernel<<<N_ROWS / 128, 256, 0, stream>>>(x, colsum);
  dim3 ggrid(N_FEAT / 64, N_FEAT / 64, N_ROWS / K_CHUNK);
  gram_wmma_kernel<<<ggrid, 128, 0, stream>>>(x, G);
  mask_kernel<<<1, 1024, (size_t)N_FEAT * 32 * sizeof(unsigned int), stream>>>(G, colsum, mask);
  scale_kernel<<<4096, 256, 0, stream>>>(x, mask, out);
}
